// CapsuleNetwork_24180665876665
// MI455X (gfx1250) — compile-verified, matrix operand order checked by reading, not run on hardware
//
#include <hip/hip_runtime.h>
#include <hip/hip_bf16.h>
#include <stdint.h>

typedef __bf16 bf16;
typedef bf16 v16bf __attribute__((ext_vector_type(16)));
typedef float v8f  __attribute__((ext_vector_type(8)));

union FragU { uint4 q[2]; v16bf f; };

// Async DMA global -> LDS, 16B per lane, tracked by ASYNCcnt (CDNA5 path).
// offset:N is added to both the LDS destination and the global source address.
__device__ __forceinline__ void async_ld_b128(uint32_t lds_off, const void* gaddr) {
    asm volatile("global_load_async_to_lds_b128 %0, %1, off"
                 :: "v"(lds_off), "v"((uint64_t)(uintptr_t)gaddr) : "memory");
}
__device__ __forceinline__ void async_ld_b128_o16(uint32_t lds_off, const void* gaddr) {
    asm volatile("global_load_async_to_lds_b128 %0, %1, off offset:16"
                 :: "v"(lds_off), "v"((uint64_t)(uintptr_t)gaddr) : "memory");
}
// Low 32 bits of a generic pointer to __shared__ == LDS byte offset (flat aperture).
__device__ __forceinline__ uint32_t lds_off_of(const void* p) {
    return (uint32_t)(uintptr_t)p;
}

// Problem-wide constants (all three GEMMs are 32768 x 1024 x 1024)
constexpr int GM = 32768, GN = 1024, GK = 1024;

// ---------------------------------------------------------------------------
// float -> bf16 conversion
// ---------------------------------------------------------------------------
__global__ __launch_bounds__(256) void cvt_bf16_kernel(const float* __restrict__ in,
                                                       bf16* __restrict__ out, size_t n) {
    size_t i = (size_t)blockIdx.x * 256u + threadIdx.x;
    if (i < n) out[i] = (bf16)in[i];
}

// ---------------------------------------------------------------------------
// bf16 GEMM: C[M,N] = A[M,K] * B[N,K]^T (+bias)(+relu), f32 accumulate (WMMA)
// Block tile 128x128, 8 wave32s as 4(M) x 2(N), 32x64 per wave (2x4 WMMA tiles).
// Double-buffered LDS fed by global_load_async_to_lds_b128 (ASYNCcnt pipeline):
// issue tile k+1, s_wait_asynccnt 4 => tile k resident (async loads retire in
// order). K loop kept at unroll 1 so only one fragment set is live (no acc
// register aliasing -> no v_mov/v_nop shuffle in the hot loop).
// flags: 1 = add bias, 2 = relu, 4 = bf16 output (else f32)
// ---------------------------------------------------------------------------
__global__ __launch_bounds__(256) void gemm_bf16_kernel(
    const bf16* __restrict__ A, const bf16* __restrict__ Bm,
    const float* __restrict__ bias, void* __restrict__ Cout, int flags)
{
    constexpr int BM = 128, BN = 128, BK = 32, LW = 40; // 80B LDS row stride: bank-conflict-free
    constexpr int NK = GK / BK;                         // 32 K-steps
    __shared__ __align__(16) bf16 As[2][BM * LW];
    __shared__ __align__(16) bf16 Bs[2][BN * LW];

    const int t    = threadIdx.x;
    const int lane = t & 31;
    const int wave = t >> 5;
    const int wm   = wave & 3;   // 4 waves along M (strips of 32)
    const int wn   = wave >> 2;  // 2 waves along N (strips of 64)
    const int m16  = lane & 15;
    const int half = lane >> 4;

    const int rowBlock = blockIdx.y * BM;
    const int colBlock = blockIdx.x * BN;

    // Cooperative staging: each thread DMAs 32B of A and 32B of B per K-step.
    const int ldr = t >> 1;          // 0..127
    const int ldc = (t & 1) * 16;    // 0 or 16 (bf16 cols)

    const bf16* gA = A  + (size_t)(rowBlock + ldr) * GK + ldc;
    const bf16* gB = Bm + (size_t)(colBlock + ldr) * GK + ldc;
    const uint32_t lA0 = lds_off_of(&As[0][ldr * LW + ldc]);
    const uint32_t lB0 = lds_off_of(&Bs[0][ldr * LW + ldc]);
    const uint32_t lA1 = lds_off_of(&As[1][ldr * LW + ldc]);
    const uint32_t lB1 = lds_off_of(&Bs[1][ldr * LW + ldc]);

    // Per-wave fragment read bases for the two LDS buffers
    const uint4* rA0 = (const uint4*)As[0];
    const uint4* rA1 = (const uint4*)As[1];
    const uint4* rB0 = (const uint4*)Bs[0];
    const uint4* rB1 = (const uint4*)Bs[1];

    v8f acc[2][4] = {};

    // One K-step of WMMA work from the given LDS fragment bases
    auto compute = [&](const uint4* lA, const uint4* lB) {
        FragU a[2], b[4];
#pragma unroll
        for (int i = 0; i < 2; ++i) {
            int r = wm * 32 + i * 16 + m16;
            a[i].q[0] = lA[r * 5 + half];
            a[i].q[1] = lA[r * 5 + half + 2];
        }
#pragma unroll
        for (int j = 0; j < 4; ++j) {
            int r = wn * 64 + j * 16 + m16;
            b[j].q[0] = lB[r * 5 + half];
            b[j].q[1] = lB[r * 5 + half + 2];
        }
#pragma unroll
        for (int i = 0; i < 2; ++i)
#pragma unroll
            for (int j = 0; j < 4; ++j)
                acc[i][j] = __builtin_amdgcn_wmma_f32_16x16x32_bf16(
                    false, a[i].f, false, b[j].f, (short)0, acc[i][j], false, false);
    };

    // Prologue: DMA tile 0 into buffer 0
    async_ld_b128    (lA0, gA);
    async_ld_b128_o16(lA0, gA);
    async_ld_b128    (lB0, gB);
    async_ld_b128_o16(lB0, gB);

    // Main loop: always prefetch tile it+1, then consume tile it
#pragma unroll 1
    for (int it = 0; it < NK - 1; ++it) {
        const int buf = it & 1;
        const bf16* nA = gA + (size_t)(it + 1) * BK;
        const bf16* nB = gB + (size_t)(it + 1) * BK;
        const uint32_t wa = buf ? lA0 : lA1;     // write into the *other* buffer
        const uint32_t wb = buf ? lB0 : lB1;
        async_ld_b128    (wa, nA);
        async_ld_b128_o16(wa, nA);
        async_ld_b128    (wb, nB);
        async_ld_b128_o16(wb, nB);
        asm volatile("s_wait_asynccnt 0x4" ::: "memory");  // tile `it` resident
        __syncthreads();                                   // all waves' DMA visible
        compute(buf ? rA1 : rA0, buf ? rB1 : rB0);
        __syncthreads();                                   // buffer reusable
    }
    // Epilogue K-step
    asm volatile("s_wait_asynccnt 0x0" ::: "memory");
    __syncthreads();
    {
        const int buf = (NK - 1) & 1;
        compute(buf ? rA1 : rA0, buf ? rB1 : rB0);
    }

    const bool has_bias = (flags & 1) != 0;
    const bool do_relu  = (flags & 2) != 0;
    const bool out_bf16 = (flags & 4) != 0;

    // C/D layout: lanes 0-15 -> N=lane, M=vgpr; lanes 16-31 -> N=lane-16, M=vgpr+8
#pragma unroll
    for (int i = 0; i < 2; ++i) {
#pragma unroll
        for (int j = 0; j < 4; ++j) {
            int col = colBlock + wn * 64 + j * 16 + m16;
            float bv = has_bias ? bias[col] : 0.0f;
#pragma unroll
            for (int r = 0; r < 8; ++r) {
                int row = rowBlock + wm * 32 + i * 16 + r + 8 * half;
                float val = acc[i][j][r] + bv;
                if (do_relu) val = fmaxf(val, 0.0f);
                if (out_bf16) ((bf16*)Cout)[(size_t)row * GN + col] = (bf16)val;
                else          ((float*)Cout)[(size_t)row * GN + col] = val;
            }
        }
    }
}

// ---------------------------------------------------------------------------
// Dynamic routing (IN_CAPS==1 closed form), in-place u_hat -> v
// ---------------------------------------------------------------------------
__global__ __launch_bounds__(256) void routing_kernel(float* __restrict__ uv) {
    const int b = blockIdx.x;
    const int t = threadIdx.x;
    float* row = uv + (size_t)b * 1024;

    float u[4];
#pragma unroll
    for (int n = 0; n < 4; ++n) u[n] = row[n * 256 + t];

    __shared__ float red[256];
    float usq[4];
#pragma unroll
    for (int n = 0; n < 4; ++n) {
        red[t] = u[n] * u[n];
        __syncthreads();
        for (int s = 128; s > 0; s >>= 1) {
            if (t < s) red[t] += red[t + s];
            __syncthreads();
        }
        usq[n] = red[0];
        __syncthreads();
    }

    float blog[4] = {0.f, 0.f, 0.f, 0.f};
    float factor[4];
    for (int iter = 0; iter < 3; ++iter) {
        float mx = fmaxf(fmaxf(blog[0], blog[1]), fmaxf(blog[2], blog[3]));
        float e[4], se = 0.f;
#pragma unroll
        for (int n = 0; n < 4; ++n) { e[n] = __expf(blog[n] - mx); se += e[n]; }
        float inv_se = 1.0f / se;
#pragma unroll
        for (int n = 0; n < 4; ++n) {
            float c     = e[n] * inv_se;
            float sq    = c * c * usq[n];              // ||s||^2 = c^2 * ||u||^2
            float scale = sq / (1.0f + sq);
            float f     = scale * c * rsqrtf(sq + 1e-11f); // v = f * u
            factor[n]   = f;
            if (iter < 2) blog[n] += f * usq[n];       // u.v = f * ||u||^2
        }
    }
#pragma unroll
    for (int n = 0; n < 4; ++n) row[n * 256 + t] = factor[n] * u[n];
}

// ---------------------------------------------------------------------------
// BN statistics: deterministic two-stage reduction over (B*4) rows x 256 ch.
// ---------------------------------------------------------------------------
__global__ __launch_bounds__(256) void stats_kernel(const float* __restrict__ v,
                                                    float* __restrict__ psum,
                                                    float* __restrict__ psq, int rows) {
    const int t = threadIdx.x, g = blockIdx.x, nb = gridDim.x;
    float s = 0.f, q = 0.f;
    for (int r = g; r < rows; r += nb) {
        float x = v[(size_t)r * 256 + t];
        s += x; q += x * x;
    }
    psum[g * 256 + t] = s;
    psq [g * 256 + t] = q;
}

__global__ __launch_bounds__(256) void finalize_kernel(const float* __restrict__ psum,
                                                       const float* __restrict__ psq,
                                                       const float* __restrict__ gamma,
                                                       const float* __restrict__ beta,
                                                       float* __restrict__ ab,
                                                       int nblocks, float invCount) {
    const int t = threadIdx.x;
    float s = 0.f, q = 0.f;
    for (int g = 0; g < nblocks; ++g) { s += psum[g * 256 + t]; q += psq[g * 256 + t]; }
    float mean = s * invCount;
    float var  = q * invCount - mean * mean;
    float istd = rsqrtf(var + 1e-5f);
    float a    = gamma[t] * istd;
    ab[t]       = a;                     // scale
    ab[256 + t] = beta[t] - mean * a;    // shift
}

__global__ __launch_bounds__(256) void bn_apply_kernel(const float* __restrict__ v,
                                                       const float* __restrict__ ab,
                                                       bf16* __restrict__ out, size_t n) {
    size_t i = (size_t)blockIdx.x * 256u + threadIdx.x;
    if (i < n) {
        int ch = (int)(i & 255u);
        out[i] = (bf16)(v[i] * ab[ch] + ab[256 + ch]);
    }
}

// ---------------------------------------------------------------------------
// Host-side orchestration
// ---------------------------------------------------------------------------
extern "C" void kernel_launch(void* const* d_in, const int* in_sizes, int n_in,
                              void* d_out, int out_size, void* d_ws, size_t ws_size,
                              hipStream_t stream) {
    (void)in_sizes; (void)n_in; (void)out_size; (void)ws_size;

    const float* x     = (const float*)d_in[0];
    const float* W     = (const float*)d_in[1];   // (4,1,256,1024) == row-major (1024,1024)
    const float* gamma = (const float*)d_in[2];
    const float* beta  = (const float*)d_in[3];
    const float* W1    = (const float*)d_in[4];
    const float* b1    = (const float*)d_in[5];
    const float* W2    = (const float*)d_in[6];
    const float* b2    = (const float*)d_in[7];
    float* out = (float*)d_out;

    const size_t BE = (size_t)GM * GN;            // 33.5M elements

    // Workspace layout (bytes)
    char* ws = (char*)d_ws;
    float* regA    = (float*)(ws);                           // 128MB: u_hat -> v -> h1(bf16)
    bf16*  regC    = (bf16*)(ws + (size_t)134217728);        //  64MB: x_bf16 -> h_bn_bf16
    bf16*  wFlat16 = (bf16*)(ws + (size_t)201326592);        //   2MB
    bf16*  w1_16   = (bf16*)(ws + (size_t)201326592 + 2097152);
    bf16*  w2_16   = (bf16*)(ws + (size_t)201326592 + 4194304);
    float* psum    = (float*)(ws + (size_t)207618048);       // 256KB
    float* psq     = (float*)(ws + (size_t)207618048 + 262144);
    float* ab      = (float*)(ws + (size_t)207618048 + 524288);

    // 1) Cast inputs to bf16
    cvt_bf16_kernel<<<(unsigned)((BE + 255) / 256), 256, 0, stream>>>(x, regC, BE);
    cvt_bf16_kernel<<<4096, 256, 0, stream>>>(W,  wFlat16, (size_t)GN * GK);
    cvt_bf16_kernel<<<4096, 256, 0, stream>>>(W1, w1_16,   (size_t)GN * GK);
    cvt_bf16_kernel<<<4096, 256, 0, stream>>>(W2, w2_16,   (size_t)GN * GK);

    dim3 gemmGrid(GN / 128, GM / 128);   // (8, 256)

    // 2) u_hat = x @ Wflat^T  (f32 out into regA)
    gemm_bf16_kernel<<<gemmGrid, 256, 0, stream>>>(regC, wFlat16, nullptr, regA, /*flags=*/0);

    // 3) Dynamic routing, in place (regA: u_hat -> v)
    routing_kernel<<<GM, 256, 0, stream>>>(regA);

    // 4) BN statistics (deterministic two-stage)
    stats_kernel<<<256, 256, 0, stream>>>(regA, psum, psq, GM * 4);
    finalize_kernel<<<1, 256, 0, stream>>>(psum, psq, gamma, beta, ab, 256,
                                           1.0f / (float)(GM * 4));

    // 5) Apply BN, emit bf16 h into regC (x no longer needed)
    bn_apply_kernel<<<(unsigned)((BE + 255) / 256), 256, 0, stream>>>(regA, ab, regC, BE);

    // 6) h1 = relu(h @ W1^T + b1), bf16 out into regA (v no longer needed)
    gemm_bf16_kernel<<<gemmGrid, 256, 0, stream>>>(regC, w1_16, b1, (void*)regA,
                                                   /*flags=*/1 | 2 | 4);

    // 7) out = h1 @ W2^T + b2, f32 out to d_out
    gemm_bf16_kernel<<<gemmGrid, 256, 0, stream>>>((const bf16*)regA, w2_16, b2, (void*)out,
                                                   /*flags=*/1);
}
